// TripletTransformerIncoming_47124381172179
// MI455X (gfx1250) — matrix operand order repaired and checked by reading, not compiled, not run on hardware
//
#include <hip/hip_runtime.h>
#include <cmath>

#define DDIM 128
#define HHEADS 8
#define KINC 16
#define QSCALE 0.08838834764831845f
#define GELU_C 0.7978845608028654f

typedef __attribute__((ext_vector_type(16))) __bf16 v16bf;
typedef __attribute__((ext_vector_type(8)))  float  v8f;

static __device__ __forceinline__ __bf16 f2bf(float f) {
    unsigned u = __builtin_bit_cast(unsigned, f);
    u += 0x7fffu + ((u >> 16) & 1u);            // round-to-nearest-even
    unsigned short s = (unsigned short)(u >> 16);
    return __builtin_bit_cast(__bf16, s);
}

static __device__ __forceinline__ v8f wmma_bf16(v16bf a, v16bf b, v8f c) {
    return __builtin_amdgcn_wmma_f32_16x16x32_bf16(false, a, false, b, (short)0, c, false, false);
}

// Swizzled-fragment helpers -------------------------------------------------
// A-fragment (16x32 bf16): element (m, kk) lives at lane = m + 16*((kk>>3)&1),
//                          i = (kk&7) + 8*(kk>>4)
static __device__ __forceinline__ int a_lane(int m, int kk) { return m + (((kk >> 3) & 1) << 4); }
static __device__ __forceinline__ int a_elem(int kk)        { return (kk & 7) + ((kk >> 4) << 3); }

// B-fragment (32x16 bf16): element (kk, n) lives at lane = (n&15) + 16*((kk>>3)&1),
//                          i = (kk&7) + 8*(kk>>4)
// Pre-swizzled weights: frag (ctile, kt) of W[K x N] at ((ctile*ktiles + kt)*32 + lane)*16 + i
static __device__ __forceinline__ v16bf load_b_sw(const __bf16* Wsw, int ktiles, int ctile, int kt, int lane) {
    return *(const v16bf*)(Wsw + ((((size_t)(ctile * ktiles + kt)) * 32 + lane) << 4));
}

// ---------------------------------------------------------------------------
// Kernel 0: convert f32 weights -> bf16, scattered into WMMA B-fragment layout
// ---------------------------------------------------------------------------
__global__ __launch_bounds__(256) void swizzle_w_kernel(
    const float* __restrict__ W, __bf16* __restrict__ out, int Kdim, int Ndim)
{
    int idx = blockIdx.x * 256 + threadIdx.x;
    if (idx >= Kdim * Ndim) return;
    int k = idx / Ndim, n = idx - k * Ndim;
    int kt = k >> 5, kk = k & 31;
    int ct = n >> 4;
    int lane = (n & 15) + (((kk >> 3) & 1) << 4);
    int ktiles = Kdim >> 5;
    out[((((size_t)(ct * ktiles + kt)) * 32 + lane) << 4) + ((kk & 7) + ((kk >> 4) << 3))] = f2bf(W[idx]);
}

// ---------------------------------------------------------------------------
// Kernel 1: x = LN(triplet_h); qkv = x @ qkv_w + qkv_b (q pre-scaled)
// One block = 16-row strip. 8 waves x 3 column tiles = 24 tiles of 384 cols.
// ---------------------------------------------------------------------------
__global__ __launch_bounds__(256) void ln_qkv_kernel(
    const float* __restrict__ xin, const float* __restrict__ g, const float* __restrict__ bta,
    const __bf16* __restrict__ Wsw, const float* __restrict__ bias, float* __restrict__ qkv)
{
    __shared__ v16bf ldsA[4][32];          // 4 K-tiles of swizzled A fragments
    __shared__ float red0[16][16];
    __shared__ float red1[16][16];
    __shared__ float statm[16];
    __shared__ float statr[16];

    int tid = threadIdx.x;
    int row = tid >> 4;
    int j   = tid & 15;
    int grow = blockIdx.x * 16 + row;

    float vals[8];
    float s = 0.f, ss = 0.f;
#pragma unroll
    for (int t = 0; t < 8; ++t) {
        float v = xin[(size_t)grow * DDIM + j + t * 16];
        vals[t] = v; s += v; ss += v * v;
    }
    red0[row][j] = s; red1[row][j] = ss;
    __syncthreads();
    if (j == 0) {
        float m = 0.f, m2 = 0.f;
#pragma unroll
        for (int t = 0; t < 16; ++t) { m += red0[row][t]; m2 += red1[row][t]; }
        m *= (1.f / DDIM); m2 *= (1.f / DDIM);
        statm[row] = m;
        statr[row] = rsqrtf(m2 - m * m + 1e-5f);
    }
    __syncthreads();
    float mean = statm[row], rstd = statr[row];
#pragma unroll
    for (int t = 0; t < 8; ++t) {
        int k = j + t * 16;
        float xn = (vals[t] - mean) * rstd * g[k] + bta[k];
        ((__bf16*)&ldsA[k >> 5][a_lane(row, k & 31)])[a_elem(k & 31)] = f2bf(xn);
    }
    __syncthreads();

    int wv = tid >> 5, lane = tid & 31;
#pragma unroll
    for (int t = 0; t < 3; ++t) {
        int ctile = wv * 3 + t;
        v8f c = {};
#pragma unroll
        for (int kt = 0; kt < 4; ++kt) {
            c = wmma_bf16(ldsA[kt][lane], load_b_sw(Wsw, 4, ctile, kt, lane), c);
        }
        int n = ctile * 16 + (lane & 15);
        int half = lane >> 4;
        float bv = bias[n];
        float sc = (n < DDIM) ? QSCALE : 1.0f;
#pragma unroll
        for (int r = 0; r < 8; ++r) {
            int gr = blockIdx.x * 16 + r + 8 * half;
            qkv[(size_t)gr * (3 * DDIM) + n] = (c[r] + bv) * sc;
        }
    }
}

// ---------------------------------------------------------------------------
// Kernel 2: per-(edge,head) score = dot(q[se], k[de]) + edge_bias, masked
// ---------------------------------------------------------------------------
__global__ __launch_bounds__(256) void edge_score_kernel(
    const float* __restrict__ qkv, const int* __restrict__ src, const int* __restrict__ dst,
    const float* __restrict__ eb, const unsigned char* __restrict__ mask_e,
    float* __restrict__ scores, int EH)
{
    int gid = blockIdx.x * blockDim.x + threadIdx.x;
    if (gid >= EH) return;
    int e = gid >> 3;
    int h = gid & 7;
    bool me = mask_e[e] != 0;
    int se = me ? src[e] : 0;
    int de = me ? dst[e] : 0;
    const float4* qp = (const float4*)(qkv + (size_t)se * (3 * DDIM) + h * 16);
    const float4* kp = (const float4*)(qkv + (size_t)de * (3 * DDIM) + DDIM + h * 16);
    float dot = 0.f;
#pragma unroll
    for (int t = 0; t < 4; ++t) {
        float4 a = qp[t], c = kp[t];
        dot += a.x * c.x + a.y * c.y + a.z * c.z + a.w * c.w;
    }
    scores[gid] = me ? (dot + eb[gid]) : 0.f;
}

// ---------------------------------------------------------------------------
// Kernel 3: per-node softmax over K=16 incident edges, weighted v gather-sum.
// One wave per node; 8 nodes per block.
// ---------------------------------------------------------------------------
__global__ __launch_bounds__(256) void aggregate_kernel(
    const float* __restrict__ qkv, const float* __restrict__ scores,
    const int* __restrict__ src, const unsigned char* __restrict__ mask_e,
    const int* __restrict__ inc_idx, const unsigned char* __restrict__ inc_mask,
    const unsigned char* __restrict__ mask_nodes, float* __restrict__ attn)
{
    __shared__ int   s_src[8][KINC];
    __shared__ float s_vm[8][KINC];
    __shared__ float s_a[8][KINC][HHEADS];

    int tid = threadIdx.x;
    int wv = tid >> 5, lane = tid & 31;
    int n = blockIdx.x * 8 + wv;
    bool mn = mask_nodes[n] != 0;

    if (lane < KINC) {
        int idx = inc_idx[n * KINC + lane];
        if (idx < 0) idx = 0;
        bool me = mask_e[idx] != 0;
        s_src[wv][lane] = me ? src[idx] : 0;
        s_vm[wv][lane]  = me ? 1.f : 0.f;
        bool m = (inc_mask[n * KINC + lane] != 0) && mn;
#pragma unroll
        for (int h = 0; h < HHEADS; ++h) {
            float s = scores[(size_t)idx * HHEADS + h];
            s_a[wv][lane][h] = m ? s : -1e9f;
        }
    }
    __syncthreads();
    if (lane < HHEADS) {
        int h = lane;
        float mx = -1e30f;
#pragma unroll
        for (int k = 0; k < KINC; ++k) mx = fmaxf(mx, s_a[wv][k][h]);
        float ex[KINC];
        float sum = 0.f;
#pragma unroll
        for (int k = 0; k < KINC; ++k) { ex[k] = expf(s_a[wv][k][h] - mx); sum += ex[k]; }
        float inv = 1.f / sum;
#pragma unroll
        for (int k = 0; k < KINC; ++k) s_a[wv][k][h] = ex[k] * inv;
    }
    __syncthreads();

    int d0 = lane * 4;
    int h  = lane >> 2;
    float4 acc = {0.f, 0.f, 0.f, 0.f};
#pragma unroll
    for (int k = 0; k < KINC; ++k) {
        float w = s_a[wv][k][h] * s_vm[wv][k];
        const float4 v = *(const float4*)(qkv + (size_t)s_src[wv][k] * (3 * DDIM) + 2 * DDIM + d0);
        acc.x += w * v.x; acc.y += w * v.y; acc.z += w * v.z; acc.w += w * v.w;
    }
    *(float4*)(attn + (size_t)n * DDIM + d0) = acc;
}

// ---------------------------------------------------------------------------
// Kernel 4: x2 = triplet_h + attn @ res_in_w + res_in_b  (WMMA 128x128)
// ---------------------------------------------------------------------------
__global__ __launch_bounds__(256) void res_proj_kernel(
    const float* __restrict__ attn, const float* __restrict__ x0,
    const __bf16* __restrict__ Wsw, const float* __restrict__ bias, float* __restrict__ x2)
{
    __shared__ v16bf ldsA[4][32];
    int tid = threadIdx.x;
    for (int t = tid; t < 16 * DDIM; t += 256) {
        int r = t >> 7, k = t & 127;
        float v = attn[(size_t)(blockIdx.x * 16 + r) * DDIM + k];
        ((__bf16*)&ldsA[k >> 5][a_lane(r, k & 31)])[a_elem(k & 31)] = f2bf(v);
    }
    __syncthreads();

    int wv = tid >> 5, lane = tid & 31;
    v8f c = {};
#pragma unroll
    for (int kt = 0; kt < 4; ++kt) {
        c = wmma_bf16(ldsA[kt][lane], load_b_sw(Wsw, 4, wv, kt, lane), c);
    }
    int n = wv * 16 + (lane & 15);
    int half = lane >> 4;
    float bv = bias[n];
#pragma unroll
    for (int r = 0; r < 8; ++r) {
        int gr = blockIdx.x * 16 + r + 8 * half;
        x2[(size_t)gr * DDIM + n] = x0[(size_t)gr * DDIM + n] + c[r] + bv;
    }
}

// ---------------------------------------------------------------------------
// Kernel 5: fused FFN: y = LN(x2); h = gelu(y@W1+b1) (LDS bf16); out = x2 + h@W2+b2
// ---------------------------------------------------------------------------
__global__ __launch_bounds__(256) void ffn_kernel(
    const float* __restrict__ x2, const float* __restrict__ g, const float* __restrict__ bta,
    const __bf16* __restrict__ W1sw, const float* __restrict__ b1,
    const __bf16* __restrict__ W2sw, const float* __restrict__ b2, float* __restrict__ out)
{
    __shared__ v16bf ldsA[4][32];          // LN(x2) fragments (K = 128)
    __shared__ v16bf ldsH[16][32];         // gelu hidden fragments (K = 512)
    __shared__ float red0[16][16];
    __shared__ float red1[16][16];
    __shared__ float statm[16];
    __shared__ float statr[16];

    int tid = threadIdx.x;
    int row = tid >> 4;
    int j   = tid & 15;
    int grow = blockIdx.x * 16 + row;

    float vals[8];
    float s = 0.f, ss = 0.f;
#pragma unroll
    for (int t = 0; t < 8; ++t) {
        float v = x2[(size_t)grow * DDIM + j + t * 16];
        vals[t] = v; s += v; ss += v * v;
    }
    red0[row][j] = s; red1[row][j] = ss;
    __syncthreads();
    if (j == 0) {
        float m = 0.f, m2 = 0.f;
#pragma unroll
        for (int t = 0; t < 16; ++t) { m += red0[row][t]; m2 += red1[row][t]; }
        m *= (1.f / DDIM); m2 *= (1.f / DDIM);
        statm[row] = m;
        statr[row] = rsqrtf(m2 - m * m + 1e-5f);
    }
    __syncthreads();
    float mean = statm[row], rstd = statr[row];
#pragma unroll
    for (int t = 0; t < 8; ++t) {
        int k = j + t * 16;
        float xn = (vals[t] - mean) * rstd * g[k] + bta[k];
        ((__bf16*)&ldsA[k >> 5][a_lane(row, k & 31)])[a_elem(k & 31)] = f2bf(xn);
    }
    __syncthreads();

    int wv = tid >> 5, lane = tid & 31;
    // GEMM1 (128 -> 512) + GELU into swizzled LDS fragments
#pragma unroll
    for (int t = 0; t < 4; ++t) {
        int ctile = wv * 4 + t;
        v8f c = {};
#pragma unroll
        for (int kt = 0; kt < 4; ++kt) {
            c = wmma_bf16(ldsA[kt][lane], load_b_sw(W1sw, 4, ctile, kt, lane), c);
        }
        int n = ctile * 16 + (lane & 15);
        int half = lane >> 4;
        float bv = b1[n];
        int kt2 = n >> 5, kk2 = n & 31;   // hidden dim n becomes K index of GEMM2
#pragma unroll
        for (int r = 0; r < 8; ++r) {
            int m = r + 8 * half;
            float hh = c[r] + bv;
            float u = GELU_C * (hh + 0.044715f * hh * hh * hh);
            float ge = 0.5f * hh * (1.f + tanhf(u));
            ((__bf16*)&ldsH[kt2][a_lane(m, kk2)])[a_elem(kk2)] = f2bf(ge);
        }
    }
    __syncthreads();

    // GEMM2 (512 -> 128) + residual
    v8f c = {};
#pragma unroll
    for (int kt = 0; kt < 16; ++kt) {
        c = wmma_bf16(ldsH[kt][lane], load_b_sw(W2sw, 16, wv, kt, lane), c);
    }
    int n = wv * 16 + (lane & 15);
    int half = lane >> 4;
    float bv = b2[n];
#pragma unroll
    for (int r = 0; r < 8; ++r) {
        int gr = blockIdx.x * 16 + r + 8 * half;
        out[(size_t)gr * DDIM + n] = x2[(size_t)gr * DDIM + n] + c[r] + bv;
    }
}

// ---------------------------------------------------------------------------
extern "C" void kernel_launch(void* const* d_in, const int* in_sizes, int n_in,
                              void* d_out, int out_size, void* d_ws, size_t ws_size,
                              hipStream_t stream) {
    (void)n_in; (void)out_size; (void)ws_size;
    const float*         triplet    = (const float*)d_in[0];
    const unsigned char* mask_nodes = (const unsigned char*)d_in[1];
    const int*           src        = (const int*)d_in[2];
    const int*           dst        = (const int*)d_in[3];
    const float*         edge_bias  = (const float*)d_in[4];
    const unsigned char* mask_edges = (const unsigned char*)d_in[5];
    const int*           inc_idx    = (const int*)d_in[6];
    const unsigned char* inc_mask   = (const unsigned char*)d_in[7];
    const float*         ln1_g      = (const float*)d_in[8];
    const float*         ln1_b      = (const float*)d_in[9];
    const float*         qkv_w      = (const float*)d_in[10];
    const float*         qkv_b      = (const float*)d_in[11];
    const float*         res_in_w   = (const float*)d_in[12];
    const float*         res_in_b   = (const float*)d_in[13];
    const float*         res_ln_g   = (const float*)d_in[14];
    const float*         res_ln_b   = (const float*)d_in[15];
    const float*         ffn_in_w   = (const float*)d_in[16];
    const float*         ffn_in_b   = (const float*)d_in[17];
    const float*         ffn_out_w  = (const float*)d_in[18];
    const float*         ffn_out_b  = (const float*)d_in[19];

    const int N = in_sizes[1];   // mask_nodes count
    const int E = in_sizes[2];   // src count

    float* qkv    = (float*)d_ws;
    float* scores = qkv + (size_t)N * 3 * DDIM;
    float* attn   = scores + (size_t)E * HHEADS;
    float* x2     = attn + (size_t)N * DDIM;
    __bf16* wq    = (__bf16*)(x2 + (size_t)N * DDIM);   // 128x384 = 49152
    __bf16* wr    = wq + 49152;                          // 128x128 = 16384
    __bf16* w1    = wr + 16384;                          // 128x512 = 65536
    __bf16* w2    = w1 + 65536;                          // 512x128 = 65536
    float* outp   = (float*)d_out;

    swizzle_w_kernel<<<(128 * 384 + 255) / 256, 256, 0, stream>>>(qkv_w, wq, 128, 384);
    swizzle_w_kernel<<<(128 * 128 + 255) / 256, 256, 0, stream>>>(res_in_w, wr, 128, 128);
    swizzle_w_kernel<<<(128 * 512 + 255) / 256, 256, 0, stream>>>(ffn_in_w, w1, 128, 512);
    swizzle_w_kernel<<<(512 * 128 + 255) / 256, 256, 0, stream>>>(ffn_out_w, w2, 512, 128);

    ln_qkv_kernel<<<N / 16, 256, 0, stream>>>(triplet, ln1_g, ln1_b, wq, qkv_b, qkv);
    edge_score_kernel<<<(E * HHEADS + 255) / 256, 256, 0, stream>>>(
        qkv, src, dst, edge_bias, mask_edges, scores, E * HHEADS);
    aggregate_kernel<<<N / 8, 256, 0, stream>>>(
        qkv, scores, src, mask_edges, inc_idx, inc_mask, mask_nodes, attn);
    res_proj_kernel<<<N / 16, 256, 0, stream>>>(attn, triplet, wr, res_in_b, x2);
    ffn_kernel<<<N / 16, 256, 0, stream>>>(
        x2, res_ln_g, res_ln_b, w1, ffn_in_b, w2, ffn_out_b, outp);
}
